// adaptIndexSplit_dotsim_1D_81312320848590
// MI455X (gfx1250) — matrix operand order repaired and checked by reading, not compiled, not run on hardware
//
#include <hip/hip_runtime.h>
#include <hip/hip_bf16.h>

// Problem constants
#define B_   16
#define T_   392
#define C_   512
#define E8_  64
#define K_   196
#define TT_  (T_*T_)     // 153664
#define TP_  400         // T padded to multiple of 16
#define SCALER (1.0f/512.0f)

typedef __attribute__((ext_vector_type(16))) __bf16 v16bf;
typedef __attribute__((ext_vector_type(8)))  float  v8f;
typedef __attribute__((ext_vector_type(2)))  float  v2f;

// native RNE convert (backend should emit v_cvt_*bf16_f32)
__device__ __forceinline__ __bf16 f2bf(float f) { return (__bf16)f; }

// ---------------------------------------------------------------------------
// K0: zero xi (padded rows must be 0) + convert W_dim -> bf16
// ---------------------------------------------------------------------------
__global__ void k_prep(const float* __restrict__ Wdim, __bf16* __restrict__ wdim_bf,
                       unsigned* __restrict__ xi_zero) {
  int id = blockIdx.x * blockDim.x + threadIdx.x;
  const int NXI = (B_ * TP_ * E8_) / 2;  // u32 words covering bf16 xi buffer
  const int NWD = E8_ * C_;
  if (id < NXI) xi_zero[id] = 0u;
  int w = id - NXI;
  if (w >= 0 && w < NWD) wdim_bf[w] = f2bf(Wdim[w]);
}

// ---------------------------------------------------------------------------
// K1: xi[b,t,e] = x_in[b,t,:] . W_dim[e,:] + b_dim[e]  via v_wmma_f32_16x16x32_bf16
// grid (E8/16, TP/16, B), block 32 (one wave per 16x16 tile)
// ---------------------------------------------------------------------------
__global__ __launch_bounds__(32) void k_dimproj(const float* __restrict__ x_in,
    const __bf16* __restrict__ wdim_bf, const float* __restrict__ b_dim,
    __bf16* __restrict__ xi_bf) {
  const int l = threadIdx.x;
  const int nt = blockIdx.x, mt = blockIdx.y, b = blockIdx.z;
  const int lm = l & 15, grp = l >> 4;
  const int kb = grp * 8, kb2 = grp * 16;
  int row = mt * 16 + lm;
  int rowc = row < T_ ? row : (T_ - 1);          // clamp loads for padded rows
  const float*  ar = x_in    + ((size_t)b * T_ + rowc) * C_;
  const __bf16* br = wdim_bf + (size_t)(nt * 16 + lm) * C_;   // B^T row (e, c)
  v8f acc = {};
  for (int k0 = 0; k0 < C_; k0 += 32) {
    v16bf A, Bv;
#pragma unroll
    for (int j = 0; j < 8; ++j) {
      A[j]     = f2bf(ar[k0 + kb2 + j]);         // K = kb2..kb2+7
      A[8 + j] = f2bf(ar[k0 + kb2 + 8 + j]);     // K = kb2+8..kb2+15
    }
#pragma unroll
    for (int j = 0; j < 16; ++j) Bv[j] = br[k0 + kb2 + j];    // K = kb2..kb2+15
    acc = __builtin_amdgcn_wmma_f32_16x16x32_bf16(false, A, false, Bv,
                                                  (short)0, acc, false, false);
  }
  const int e = nt * 16 + lm;
  const float bias = b_dim[e];
  __bf16* orow0 = xi_bf + ((size_t)b * TP_ + mt * 16) * E8_ + e;
  if (mt * 16 + 16 <= T_) {                      // full tile: unguarded stores
#pragma unroll
    for (int v = 0; v < 8; ++v)
      orow0[(size_t)(v + grp * 8) * E8_] = f2bf(acc[v] + bias);
  } else {
#pragma unroll
    for (int v = 0; v < 8; ++v) {
      int orow = mt * 16 + v + grp * 8;
      if (orow < T_) orow0[(size_t)(v + grp * 8) * E8_] = f2bf(acc[v] + bias);
    }
  }
}

// ---------------------------------------------------------------------------
// K2: S[b,i,j] = scaler * xi[b,i,:] . xi[b,j,:]   (bf16 WMMA, K=64 -> 2 steps)
// grid (TP/16, TP/16, B), block 32
// ---------------------------------------------------------------------------
__global__ __launch_bounds__(32) void k_attgemm(const __bf16* __restrict__ xi_bf,
                                                float* __restrict__ att) {
  const int l = threadIdx.x;
  const int nt = blockIdx.x, mt = blockIdx.y, b = blockIdx.z;
  const int lm = l & 15, grp = l >> 4;
  const int kb = grp * 8, kb2 = grp * 16;
  const __bf16* ar = xi_bf + ((size_t)b * TP_ + mt * 16 + lm) * E8_;
  const __bf16* br = xi_bf + ((size_t)b * TP_ + nt * 16 + lm) * E8_;
  v8f acc = {};
#pragma unroll
  for (int k0 = 0; k0 < E8_; k0 += 32) {
    v16bf A, Bv;
#pragma unroll
    for (int j = 0; j < 8; ++j) {
      A[j]     = ar[k0 + kb2 + j];
      A[8 + j] = ar[k0 + kb2 + 8 + j];
    }
#pragma unroll
    for (int j = 0; j < 16; ++j) Bv[j] = br[k0 + kb2 + j];
    acc = __builtin_amdgcn_wmma_f32_16x16x32_bf16(false, A, false, Bv,
                                                  (short)0, acc, false, false);
  }
  const int col = nt * 16 + lm;
  float* out0 = att + (size_t)b * TT_ + (size_t)(mt * 16) * T_ + col;
  const bool fullM = (mt * 16 + 16 <= T_);
  if (fullM && col < T_ && nt * 16 + 16 <= T_) { // interior tile: unguarded
#pragma unroll
    for (int v = 0; v < 8; ++v)
      out0[(size_t)(v + grp * 8) * T_] = acc[v] * SCALER;
  } else {
#pragma unroll
    for (int v = 0; v < 8; ++v) {
      int rrow = mt * 16 + v + grp * 8;
      if (rrow < T_ && col < T_)
        out0[(size_t)(v + grp * 8) * T_] = acc[v] * SCALER;
    }
  }
}

// ---------------------------------------------------------------------------
// K3: row softmax over j<392, in place.  grid (T, B), block 128
// ---------------------------------------------------------------------------
__global__ void k_softmax(float* __restrict__ att) {
  const int i = blockIdx.x, b = blockIdx.y, tid = threadIdx.x;
  float* row = att + (size_t)b * TT_ + (size_t)i * T_;
  __shared__ float red[128];
  float mx = -3.4e38f;
  for (int j = tid; j < T_; j += 128) mx = fmaxf(mx, row[j]);
  red[tid] = mx; __syncthreads();
  for (int s = 64; s > 0; s >>= 1) {
    if (tid < s) red[tid] = fmaxf(red[tid], red[tid + s]);
    __syncthreads();
  }
  mx = red[0]; __syncthreads();
  float sum = 0.f;
  for (int j = tid; j < T_; j += 128) { float e = expf(row[j] - mx); row[j] = e; sum += e; }
  red[tid] = sum; __syncthreads();
  for (int s = 64; s > 0; s >>= 1) {
    if (tid < s) red[tid] += red[tid + s];
    __syncthreads();
  }
  float inv = 1.0f / red[0];
  for (int j = tid; j < T_; j += 128) row[j] *= inv;
}

// ---------------------------------------------------------------------------
// K4: the 241MB streaming GEMM  xt[m,n] = att_flat[m,:] . W_time[n,:]
// M=16 (batch) single tile, fp32 WMMA 16x16x4, 5 N-tiles per wave (A reuse),
// K split into KS chunks; each wave writes a deterministic partial block,
// K5r reduces them (no atomics).   grid (TP/80, KS), block 32
// ---------------------------------------------------------------------------
#define NT_ 5
#define KS_ 256
__global__ __launch_bounds__(32) void k_timegemm(const float* __restrict__ att,
    const float* __restrict__ W_time, float* __restrict__ part) {
  const int l = threadIdx.x;
  const int lm = l & 15, grp = l >> 4, kt = grp * 2;
  const int nbase = blockIdx.x * (NT_ * 16);
  const int ky = blockIdx.y;
  const int ksteps_total = TT_ / 4;                       // 38416
  const int chunk = (ksteps_total + KS_ - 1) / KS_;       // 151
  int s0 = ky * chunk;
  int s1 = s0 + chunk; if (s1 > ksteps_total) s1 = ksteps_total;
  int nsteps = s1 - s0;

  const float* ap = att + (size_t)lm * TT_ + 4 * s0 + kt; // A row = batch lm
  const float* bp[NT_];
#pragma unroll
  for (int i = 0; i < NT_; ++i) {
    int n = nbase + i * 16 + lm;
    if (n > T_ - 1) n = T_ - 1;                           // clamp padded-col loads
    bp[i] = W_time + (size_t)n * TT_ + 4 * s0 + kt;
  }
  v8f acc[NT_];
#pragma unroll
  for (int i = 0; i < NT_; ++i) acc[i] = {};

#pragma unroll 2
  for (int s = 0; s < nsteps; ++s) {
    v2f Av; Av[0] = ap[0]; Av[1] = ap[1];
    __builtin_prefetch(ap + 512, 0, 1);
#pragma unroll
    for (int i = 0; i < NT_; ++i) {
      v2f Bv; Bv[0] = bp[i][0]; Bv[1] = bp[i][1];
      acc[i] = __builtin_amdgcn_wmma_f32_16x16x4_f32(false, Av, false, Bv,
                                                     (short)0, acc[i], false, false);
      bp[i] += 4;
    }
    ap += 4;
  }

  // deterministic partial store: part[ky][m][n], n in 0..399 (pad cols dropped later)
  float* pr = part + (size_t)ky * (16 * TP_);
#pragma unroll
  for (int i = 0; i < NT_; ++i) {
    int n = nbase + i * 16 + lm;
#pragma unroll
    for (int v = 0; v < 8; ++v) {
      int m = v + grp * 8;                                // m = batch index
      pr[(size_t)m * TP_ + n] = acc[i][v];
    }
  }
}

// K4b: xt_acc[m,n] = sum_ky part[ky][m][n]
__global__ void k_reduce(const float* __restrict__ part, float* __restrict__ xt_acc) {
  int id = blockIdx.x * blockDim.x + threadIdx.x;
  if (id >= B_ * T_) return;
  int m = id / T_, n = id % T_;
  float s = 0.f;
  for (int ky = 0; ky < KS_; ++ky) s += part[(size_t)ky * (16 * TP_) + (size_t)m * TP_ + n];
  xt_acc[id] = s;
}

// ---------------------------------------------------------------------------
// K5: off_{a,d}[b,k] = tanh( (xt_acc[b,:]+b_time) . W_off{a,d}[k,:] ) * 2
// ---------------------------------------------------------------------------
__global__ void k_off(const float* __restrict__ xt_acc, const float* __restrict__ b_time,
                      const float* __restrict__ W_offa, const float* __restrict__ W_offd,
                      float* __restrict__ off_a, float* __restrict__ off_d) {
  int id = blockIdx.x * blockDim.x + threadIdx.x;
  if (id >= 2 * B_ * K_) return;
  int which = id / (B_ * K_);
  int rem = id % (B_ * K_);
  int b = rem / K_, k2 = rem % K_;
  const float* W  = which ? W_offd : W_offa;
  const float* xr = xt_acc + b * T_;
  const float* wr = W + (size_t)k2 * T_;
  float acc = 0.f;
  for (int n = 0; n < T_; ++n) acc += (xr[n] + b_time[n]) * wr[n];
  float val = tanhf(acc) * 2.0f;
  (which ? off_d : off_a)[rem] = val;
}

// ---------------------------------------------------------------------------
// K6: bilinear 1D grid-sample along channel axis (border clamp, align_corners=F)
// out layout: [x_a (B,T,K) | x_d (B,T,K)]
// ---------------------------------------------------------------------------
__global__ void k_sample(const float* __restrict__ x_in, const float* __restrict__ off_a,
                         const float* __restrict__ off_d, float* __restrict__ out) {
  int id = blockIdx.x * blockDim.x + threadIdx.x;
  const int total = B_ * T_ * K_;
  if (id >= total) return;
  int k2 = id % K_;
  int t  = (id / K_) % T_;
  int b  = id / (K_ * T_);
  const float* xr = x_in + ((size_t)b * T_ + t) * C_;
  const float denom = (float)(T_ - 1);
#pragma unroll
  for (int which = 0; which < 2; ++which) {
    float off = (which ? off_d : off_a)[b * K_ + k2];
    float idx = (float)(2 * k2 + which) + off;
    float g   = 2.0f * idx / denom - 1.0f;
    float ix  = ((g + 1.0f) * (float)C_ - 1.0f) * 0.5f;
    float x0f = floorf(ix);
    float w   = ix - x0f;
    int x0 = (int)x0f; int x1 = x0 + 1;
    x0 = x0 < 0 ? 0 : (x0 > C_ - 1 ? C_ - 1 : x0);
    x1 = x1 < 0 ? 0 : (x1 > C_ - 1 ? C_ - 1 : x1);
    float res = xr[x0] * (1.0f - w) + xr[x1] * w;
    out[(size_t)which * total + id] = res;
  }
}

// ---------------------------------------------------------------------------
extern "C" void kernel_launch(void* const* d_in, const int* in_sizes, int n_in,
                              void* d_out, int out_size, void* d_ws, size_t ws_size,
                              hipStream_t stream) {
  const float* x_in   = (const float*)d_in[0];
  const float* W_dim  = (const float*)d_in[1];
  const float* b_dim  = (const float*)d_in[2];
  const float* W_time = (const float*)d_in[3];
  const float* b_time = (const float*)d_in[4];
  const float* W_offa = (const float*)d_in[5];
  const float* W_offd = (const float*)d_in[6];
  float* out = (float*)d_out;

  // workspace layout (bytes, all 256-aligned)
  char* ws = (char*)d_ws;
  __bf16* wdim_bf = (__bf16*)(ws + 0);          //  64*512 bf16           = 65536
  __bf16* xi_bf   = (__bf16*)(ws + 65536);      //  16*400*64 bf16        = 819200
  float*  att     = (float*) (ws + 884736);     //  16*392*392 f32        = 9834496
  float*  part    = (float*) (ws + 10719232);   //  KS*16*400 f32         = 6553600
  float*  xt_acc  = (float*) (ws + 17272832);   //  16*392 f32            = 25088
  float*  off_a   = (float*) (ws + 17297920);   //  16*196 f32            = 12544
  float*  off_d   = (float*) (ws + 17310464);   //  16*196 f32            = 12544

  {
    int n = (B_ * TP_ * E8_) / 2 + E8_ * C_;
    k_prep<<<(n + 255) / 256, 256, 0, stream>>>(W_dim, wdim_bf, (unsigned*)xi_bf);
  }
  k_dimproj<<<dim3(E8_ / 16, TP_ / 16, B_), 32, 0, stream>>>(x_in, wdim_bf, b_dim, xi_bf);
  k_attgemm<<<dim3(TP_ / 16, TP_ / 16, B_), 32, 0, stream>>>(xi_bf, att);
  k_softmax<<<dim3(T_, B_), 128, 0, stream>>>(att);
  k_timegemm<<<dim3(TP_ / (NT_ * 16), KS_), 32, 0, stream>>>(att, W_time, part);
  {
    int n = B_ * T_;
    k_reduce<<<(n + 255) / 256, 256, 0, stream>>>(part, xt_acc);
  }
  {
    int n = 2 * B_ * K_;
    k_off<<<(n + 255) / 256, 256, 0, stream>>>(xt_acc, b_time, W_offa, W_offd, off_a, off_d);
  }
  {
    int n = B_ * T_ * K_;
    k_sample<<<(n + 255) / 256, 256, 0, stream>>>(x_in, off_a, off_d, out);
  }
}
